// Attention_83958020702472
// MI455X (gfx1250) — matrix operand order repaired and checked by reading, not compiled, not run on hardware
//
#include <hip/hip_runtime.h>

typedef __attribute__((ext_vector_type(16))) __bf16        v16bf;
typedef __attribute__((ext_vector_type(8)))  float         v8f;
typedef __attribute__((ext_vector_type(4)))  unsigned int  v4u;

union bfu { v16bf v; v4u u[2]; };

// ---------------------------------------------------------------------------
// Elementwise f32 -> bf16 (row-major), 4 elements per thread.
// ---------------------------------------------------------------------------
__global__ void cvt_bf16(const float* __restrict__ src, __bf16* __restrict__ dst,
                         int n4)
{
    const int i = blockIdx.x * blockDim.x + threadIdx.x;
    if (i >= n4) return;
    const float4 f = *(const float4*)(src + (size_t)i * 4);
    __bf16* d = dst + (size_t)i * 4;
    d[0] = (__bf16)f.x; d[1] = (__bf16)f.y;
    d[2] = (__bf16)f.z; d[3] = (__bf16)f.w;
}

// ---------------------------------------------------------------------------
// Pack a row-major f32 [K,N] matrix into the WMMA bf16 B-fragment order:
//   dst[(((k>>5)*2 + ((k>>4)&1))*N + n)*16 + (k&15)] = (bf16)src[k*N + n]
// so one lane's half-fragment (16 bf16 = 32 B) is contiguous.
// Batched via blockIdx.y.
// ---------------------------------------------------------------------------
__global__ void pack_Bmat(const float* __restrict__ src, __bf16* __restrict__ dst,
                          int K, int N, long ssrc, long sdst)
{
    const int i = blockIdx.x * blockDim.x + threadIdx.x;   // i = k*N + n
    if (i >= K * N) return;
    const int k = i / N, n = i % N;
    const int kt = k >> 5, hf = (k >> 4) & 1, j = k & 15;
    const float v = src[(size_t)blockIdx.y * ssrc + i];
    dst[(size_t)blockIdx.y * sdst + ((size_t)((kt << 1) | hf) * N + n) * 16 + j] =
        (__bf16)v;
}

// ---------------------------------------------------------------------------
// Batched GEMM: Y[b] = A[b] (bf16, row-major [M,K]) * B[b] (bf16, packed) + bias.
// One wave computes a 16x32 tile (two 16x16 WMMAs sharing the A fragment);
// 8 waves per block; batch = blockIdx.y.  N, K compile-time.
// ---------------------------------------------------------------------------
template<int N, int K>
__global__ void wmma_gemm_bf16(const __bf16* __restrict__ A,
                               const __bf16* __restrict__ Bp,
                               const float* __restrict__ bias,
                               float* __restrict__ Y,
                               int M, long sab, long sbb, long syb)
{
    constexpr int TILES_N2 = N / 32;
    const int wave  = threadIdx.x >> 5;
    const int lane  = threadIdx.x & 31;
    const int gtile = blockIdx.x * 8 + wave;
    const int tm = (gtile / TILES_N2) << 4;
    const int tn = (gtile % TILES_N2) << 5;
    if (tm >= M) return;

    const int half = lane >> 4;
    const int mr   = lane & 15;
    const int nc   = lane & 15;

    const __bf16* Ab = A  + (size_t)blockIdx.y * sab;
    const __bf16* Bb = Bp + (size_t)blockIdx.y * sbb;
    float*        Yb = Y  + (size_t)blockIdx.y * syb;

    // bias hoisted: uniform branch once, no branches in the store loop
    const float bv0 = bias ? bias[tn + nc]      : 0.0f;
    const float bv1 = bias ? bias[tn + 16 + nc] : 0.0f;

    // A: lane's two contiguous 16-byte runs start at K-offsets half*16 and half*16+8
    const __bf16* Arow = Ab + (size_t)(tm + mr) * K + half * 16;
    // B packed: lane's 32 contiguous bytes at ((kt*2+half)*N + col)*16
    const __bf16* Bc0 = Bb + ((size_t)half * N + tn + nc)      * 16;
    const __bf16* Bc1 = Bb + ((size_t)half * N + tn + 16 + nc) * 16;

    v8f acc0 = {}, acc1 = {};

#pragma unroll 4
    for (int kt = 0; kt < K / 32; ++kt) {
        bfu a, b0, b1;
        a.u[0]  = *(const v4u*)(Arow + kt * 32);
        a.u[1]  = *(const v4u*)(Arow + kt * 32 + 8);
        const __bf16* pb0 = Bc0 + (size_t)kt * 2 * N * 16;
        const __bf16* pb1 = Bc1 + (size_t)kt * 2 * N * 16;
        b0.u[0] = *(const v4u*)(pb0);
        b0.u[1] = *(const v4u*)(pb0 + 8);
        b1.u[0] = *(const v4u*)(pb1);
        b1.u[1] = *(const v4u*)(pb1 + 8);
        acc0 = __builtin_amdgcn_wmma_f32_16x16x32_bf16(
                   false, a.v, false, b0.v, (short)0, acc0, false, false);
        acc1 = __builtin_amdgcn_wmma_f32_16x16x32_bf16(
                   false, a.v, false, b1.v, (short)0, acc1, false, false);
    }

    // C/D layout: VGPR r -> row tm + r + 8*half, col tn(+16) + nc
#pragma unroll
    for (int r = 0; r < 8; ++r) {
        float* yrow = Yb + (size_t)(tm + r + 8 * half) * N;
        yrow[tn + nc]      = acc0[r] + bv0;
        yrow[tn + 16 + nc] = acc1[r] + bv1;
    }
}

// ---------------------------------------------------------------------------
// Fused additive-attention scoring + softmax (TRANS-pipe bound: v_tanh_f32).
// One 256-thread block per (b,t).  Vectorized float4 path: per iteration one
// global_load_b128 + two ds_load_b128 feed four v_tanh_f32 (mutually
// independent -> satisfies the trans-op hazard rule and keeps TRANS saturated
// while the FMA chain co-executes on VALU).
// ---------------------------------------------------------------------------
__global__ void score_softmax(const float* __restrict__ W1_hs,
                              const float* __restrict__ W2_ht,
                              const float* __restrict__ V,
                              float* __restrict__ attn,
                              __bf16* __restrict__ attnb,
                              int T, int S, int E)
{
    const int bt = blockIdx.x;
    const int b  = bt / T;

    __shared__ __align__(16) float w2sh[512];
    __shared__ __align__(16) float vsh[512];
    __shared__ float logits[128];
    __shared__ float red[128];

    const float* w2row = W2_ht + (size_t)bt * E;
    for (int i = threadIdx.x; i < E; i += blockDim.x) {
        w2sh[i] = w2row[i];
        vsh[i]  = V[i];
    }
    __syncthreads();

    const int wave = threadIdx.x >> 5;
    const int lane = threadIdx.x & 31;
    const float* w1base = W1_hs + (size_t)b * S * E;
    const float4* w2v = (const float4*)w2sh;
    const float4* vv  = (const float4*)vsh;

    for (int s = wave; s < S; s += 8) {
        const float4* w1v = (const float4*)(w1base + (size_t)s * E);
        float acc = 0.0f;
#pragma unroll
        for (int i = 0; i < 4; ++i) {
            const int e4 = lane + 32 * i;          // float4 index, coalesced
            const float4 a  = w1v[e4];
            const float4 w  = w2v[e4];
            const float4 vw = vv[e4];
            float t0, t1, t2, t3;
            asm volatile("v_tanh_f32 %0, %1" : "=v"(t0) : "v"(a.x + w.x));
            asm volatile("v_tanh_f32 %0, %1" : "=v"(t1) : "v"(a.y + w.y));
            asm volatile("v_tanh_f32 %0, %1" : "=v"(t2) : "v"(a.z + w.z));
            asm volatile("v_tanh_f32 %0, %1\n\tv_nop"
                         : "=v"(t3) : "v"(a.w + w.w));
            acc = fmaf(t0, vw.x, acc);
            acc = fmaf(t1, vw.y, acc);
            acc = fmaf(t2, vw.z, acc);
            acc = fmaf(t3, vw.w, acc);
        }
#pragma unroll
        for (int off = 16; off > 0; off >>= 1)
            acc += __shfl_xor(acc, off, 32);
        if (lane == 0) logits[s] = acc;
    }
    __syncthreads();

    const float x = (threadIdx.x < S) ? logits[threadIdx.x] : -3.0e38f;
    if (threadIdx.x < 128) red[threadIdx.x] = x;
    __syncthreads();
    for (int off = 64; off > 0; off >>= 1) {
        if (threadIdx.x < off)
            red[threadIdx.x] = fmaxf(red[threadIdx.x], red[threadIdx.x + off]);
        __syncthreads();
    }
    const float mx = red[0];
    __syncthreads();
    const float ex = (threadIdx.x < S) ? __expf(x - mx) : 0.0f;
    if (threadIdx.x < 128) red[threadIdx.x] = ex;
    __syncthreads();
    for (int off = 64; off > 0; off >>= 1) {
        if (threadIdx.x < off)
            red[threadIdx.x] += red[threadIdx.x + off];
        __syncthreads();
    }
    const float inv = 1.0f / red[0];
    if (threadIdx.x < S) {
        const float w = ex * inv;
        attn [(size_t)bt * S + threadIdx.x] = w;
        attnb[(size_t)bt * S + threadIdx.x] = (__bf16)w;
    }
}

// ---------------------------------------------------------------------------
extern "C" void kernel_launch(void* const* d_in, const int* in_sizes, int n_in,
                              void* d_out, int out_size, void* d_ws, size_t ws_size,
                              hipStream_t stream)
{
    (void)in_sizes; (void)n_in; (void)out_size; (void)ws_size;
    const int B = 16, S = 128, T = 128, E = 512;
    const size_t ACT = (size_t)B * S * E;          // 1,048,576
    const size_t WEL = (size_t)E * E;              //   262,144

    const float* enc  = (const float*)d_in[0];
    const float* dec  = (const float*)d_in[1];
    const float* W1   = (const float*)d_in[2];
    const float* W2   = (const float*)d_in[3];
    const float* bias = (const float*)d_in[4];
    const float* V    = (const float*)d_in[5];

    float* ctx  = (float*)d_out;
    float* attn = (float*)d_out + ACT;

    // workspace layout
    float*  w1hs  = (float*)d_ws;                  // 4 MB
    float*  w2ht  = w1hs + ACT;                    // 4 MB
    __bf16* encA  = (__bf16*)(w2ht + ACT);         // 2 MB (A operand, row-major)
    __bf16* decA  = encA + ACT;                    // 2 MB
    __bf16* encBp = decA + ACT;                    // 2 MB (B operand, packed, per batch)
    __bf16* w1pk  = encBp + ACT;                   // 0.5 MB
    __bf16* w2pk  = w1pk + WEL;                    // 0.5 MB
    __bf16* attnb = w2pk + WEL;                    // 0.5 MB

    // --- convert activations to bf16 (row-major A operands) ---
    cvt_bf16<<<dim3((int)(ACT / 4 / 256)), 256, 0, stream>>>(enc, encA, (int)(ACT / 4));
    cvt_bf16<<<dim3((int)(ACT / 4 / 256)), 256, 0, stream>>>(dec, decA, (int)(ACT / 4));

    // --- pack B operands into WMMA fragment order ---
    pack_Bmat<<<dim3((int)(WEL / 256)), 256, 0, stream>>>(W1, w1pk, E, E, 0, 0);
    pack_Bmat<<<dim3((int)(WEL / 256)), 256, 0, stream>>>(W2, w2pk, E, E, 0, 0);
    pack_Bmat<<<dim3((int)(S * E / 256), B), 256, 0, stream>>>(
        enc, encBp, S, E, (long)S * E, (long)S * E);

    // --- GEMM 1: W1_hs = enc * W1        (M=2048, N=512, K=512) ---
    {
        const int M = B * S;
        const int waves = (M / 16) * (E / 32);     // 2048
        wmma_gemm_bf16<512, 512><<<dim3(waves / 8), 256, 0, stream>>>(
            encA, w1pk, nullptr, w1hs, M, 0, 0, 0);
    }
    // --- GEMM 2: W2_ht = dec * W2 + b ---
    {
        const int M = B * T;
        const int waves = (M / 16) * (E / 32);
        wmma_gemm_bf16<512, 512><<<dim3(waves / 8), 256, 0, stream>>>(
            decA, w2pk, bias, w2ht, M, 0, 0, 0);
    }
    // --- fused tanh scoring + softmax ---
    score_softmax<<<dim3(B * T), 256, 0, stream>>>(w1hs, w2ht, V, attn, attnb, T, S, E);

    // --- context[b] = attn[b] (128x128) * enc[b] (128x512), batched ---
    {
        const int waves = (T / 16) * (E / 32);     // 128 per batch
        wmma_gemm_bf16<512, 128><<<dim3(waves / 8, B), 256, 0, stream>>>(
            attnb, encBp, nullptr, ctx, T,
            (long)T * S, (long)S * E, (long)T * E);
    }
}
